// VectorQuantizer_85100482003633
// MI455X (gfx1250) — matrix operand order repaired and checked by reading, not compiled, not run on hardware
//
#include <hip/hip_runtime.h>
#include <stdint.h>

#define D 256
#define K 8192
#define NROWS 32768           // B*H*W = 32*32*32
#define TOTAL (NROWS * D)     // 8388608 output elements
#define KC 64                 // codes staged per LDS tile
#define BSTRIDE 264           // padded LDS stride in bf16 units (528B = 132 dwords; 132%64=4)

typedef __attribute__((ext_vector_type(16))) __bf16        v16bf;
typedef __attribute__((ext_vector_type(8)))  float          v8f;
typedef __attribute__((ext_vector_type(8)))  unsigned short ush8;

union Frag { v16bf bf; ush8 h[2]; };

__device__ __forceinline__ unsigned short f2bf(float x) {
    unsigned int u = __float_as_uint(x);
    unsigned int r = u + 0x7FFFu + ((u >> 16) & 1u);   // round-to-nearest-even
    return (unsigned short)(r >> 16);
}

// latents [B, D, H, W] f32  ->  A [N, D] bf16 with n = b*1024 + h*32 + w
__global__ void prep_latents(const float* __restrict__ lat, unsigned short* __restrict__ A) {
    int g   = blockIdx.x * 256 + threadIdx.x;     // flat index over [B, D, H, W]
    int b   = g >> 18;
    int rem = g & ((1 << 18) - 1);
    int d   = rem >> 10;
    int hw  = rem & 1023;
    int n   = (b << 10) + hw;
    A[n * D + d] = f2bf(lat[g]);
}

// embeddings [D, K] f32 -> ET [K, D] bf16 and e_norm[k] = sum_d e^2 (f32)
__global__ void prep_embed(const float* __restrict__ E, unsigned short* __restrict__ ET,
                           float* __restrict__ enorm) {
    int k = blockIdx.x;
    int d = threadIdx.x;
    float v = E[d * K + k];
    ET[k * D + d] = f2bf(v);
    float s = v * v;
    #pragma unroll
    for (int m = 16; m >= 1; m >>= 1) s += __shfl_xor(s, m, 32);
    __shared__ float ps[8];
    if ((d & 31) == 0) ps[d >> 5] = s;
    __syncthreads();
    if (d == 0) {
        float t = 0.f;
        #pragma unroll
        for (int i = 0; i < 8; ++i) t += ps[i];
        enorm[k] = t;
    }
}

struct SMem {
    unsigned short B[2][KC * BSTRIDE];   // double-buffered codebook tile (bf16 bits)
    float          En[2][KC];            // double-buffered ||e||^2 tile
};

// Fused GEMM (bf16 WMMA) + argmin over K. Block = 256 thr = 8 waves, each wave owns 32 rows.
__global__ void __launch_bounds__(256, 1) argmin_kernel(
    const unsigned short* __restrict__ A,
    const unsigned short* __restrict__ ET,
    const float* __restrict__ enorm,
    int* __restrict__ idxOut) {

    __shared__ SMem sm;

    const int tid     = threadIdx.x;
    const int lane    = tid & 31;
    const int wid     = tid >> 5;
    const int half    = lane >> 4;
    const int l15     = lane & 15;
    const int rowBase = blockIdx.x * 256 + wid * 32;

    // ---- per-thread async staging of one 64-code tile into LDS buffer `buf` ----
    auto stage = [&](int c0, int buf) {
        #pragma unroll
        for (int p = 0; p < 8; ++p) {               // 2048 x 16B segments, coalesced
            int seg = p * 256 + tid;
            int cr  = seg >> 5;                     // code within tile
            int dp  = (seg & 31) << 3;              // d position
            unsigned           l = (unsigned)(uintptr_t)&sm.B[buf][cr * BSTRIDE + dp];
            unsigned long long g = (unsigned long long)(ET + (size_t)(c0 + cr) * D + dp);
            asm volatile("global_load_async_to_lds_b128 %0, %1, off"
                         :: "v"(l), "v"(g) : "memory");
        }
        if (tid < KC) {
            unsigned           l = (unsigned)(uintptr_t)&sm.En[buf][tid];
            unsigned long long g = (unsigned long long)(enorm + c0 + tid);
            asm volatile("global_load_async_to_lds_b32 %0, %1, off"
                         :: "v"(l), "v"(g) : "memory");
        }
    };

    // Two 16-row A tiles per wave, resident in registers for the whole K loop.
    Frag a0[8], a1[8];
    const unsigned short* arow0 = A + (size_t)(rowBase + l15) * D;
    const unsigned short* arow1 = A + (size_t)(rowBase + 16 + l15) * D;
    #pragma unroll
    for (int ds = 0; ds < 8; ++ds) {
        int o = ds * 32 + 16 * half;                // A layout: d = {o..o+15} (contiguous, matches B)
        a0[ds].h[0] = *(const ush8*)(arow0 + o);
        a0[ds].h[1] = *(const ush8*)(arow0 + o + 8);
        a1[ds].h[0] = *(const ush8*)(arow1 + o);
        a1[ds].h[1] = *(const ush8*)(arow1 + o + 8);
    }

    float mv0[8], mv1[8];
    int   mi0[8], mi1[8];
    #pragma unroll
    for (int v = 0; v < 8; ++v) { mv0[v] = 3.4e38f; mv1[v] = 3.4e38f; mi0[v] = 0; mi1[v] = 0; }

    // Prologue: fill buffer 0 with chunk 0.
    stage(0, 0);
    asm volatile("s_wait_asynccnt 0x0" ::: "memory");
    __syncthreads();

    int cur = 0;
    for (int c0 = 0; c0 < K; c0 += KC) {
        int nxt = cur ^ 1;
        if (c0 + KC < K) stage(c0 + KC, nxt);       // DMA next tile while computing this one

        const unsigned short* bbase = sm.B[cur];
        // 2-deep pipelined B-fragment fetch so ds_loads issue ahead of the consuming WMMAs.
        Frag b[2];
        {
            const unsigned short* p0 = bbase + (size_t)l15 * BSTRIDE + 16 * half;
            b[0].h[0] = *(const ush8*)(p0);
            b[0].h[1] = *(const ush8*)(p0 + 8);
        }
        #pragma unroll
        for (int sub = 0; sub < 4; ++sub) {
            v8f acc0 = {0.f, 0.f, 0.f, 0.f, 0.f, 0.f, 0.f, 0.f};
            v8f acc1 = {0.f, 0.f, 0.f, 0.f, 0.f, 0.f, 0.f, 0.f};
            #pragma unroll
            for (int ds = 0; ds < 8; ++ds) {
                int t = sub * 8 + ds;
                if (t < 31) {                        // prefetch fragment t+1
                    int ns = (t + 1) >> 3, nd = (t + 1) & 7;
                    const unsigned short* p =
                        bbase + (ns * 16 + l15) * BSTRIDE + 16 * half + nd * 32;
                    b[(t + 1) & 1].h[0] = *(const ush8*)(p);
                    b[(t + 1) & 1].h[1] = *(const ush8*)(p + 8);
                }
                acc0 = __builtin_amdgcn_wmma_f32_16x16x32_bf16(
                           false, a0[ds].bf, false, b[t & 1].bf, (short)0, acc0, false, false);
                acc1 = __builtin_amdgcn_wmma_f32_16x16x32_bf16(
                           false, a1[ds].bf, false, b[t & 1].bf, (short)0, acc1, false, false);
            }
            float en = sm.En[cur][sub * 16 + l15];
            int   cg = c0 + sub * 16 + l15;
            #pragma unroll
            for (int v = 0; v < 8; ++v) {
                float d0 = en - 2.0f * acc0[v];
                float d1 = en - 2.0f * acc1[v];
                if (d0 < mv0[v]) { mv0[v] = d0; mi0[v] = cg; }
                if (d1 < mv1[v]) { mv1[v] = d1; mi1[v] = cg; }
            }
        }

        asm volatile("s_wait_asynccnt 0x0" ::: "memory");   // DMA for `nxt` complete
        __syncthreads();                                    // all waves done with cur/nxt
        cur = nxt;
    }

    // Reduce (min,argmin) across the 16 column-lanes of each half-group.
    #pragma unroll
    for (int v = 0; v < 8; ++v) {
        #pragma unroll
        for (int m = 8; m >= 1; m >>= 1) {
            float o0 = __shfl_xor(mv0[v], m, 32); int j0 = __shfl_xor(mi0[v], m, 32);
            if (o0 < mv0[v] || (o0 == mv0[v] && j0 < mi0[v])) { mv0[v] = o0; mi0[v] = j0; }
            float o1 = __shfl_xor(mv1[v], m, 32); int j1 = __shfl_xor(mi1[v], m, 32);
            if (o1 < mv1[v] || (o1 == mv1[v] && j1 < mi1[v])) { mv1[v] = o1; mi1[v] = j1; }
        }
        if (l15 == 0) {                              // C layout: VGPR v -> row v + 8*half
            idxOut[rowBase + half * 8 + v]      = mi0[v];
            idxOut[rowBase + 16 + half * 8 + v] = mi1[v];
        }
    }
}

// Gather exact f32 codebook vectors, write [B,D,H,W] output, accumulate loss.
__global__ void gather_loss(const float* __restrict__ E, const float* __restrict__ lat,
                            const int* __restrict__ idx, float* __restrict__ out,
                            float* __restrict__ loss) {
    int g   = blockIdx.x * 256 + threadIdx.x;
    int b   = g >> 18;
    int rem = g & ((1 << 18) - 1);
    int d   = rem >> 10;
    int hw  = rem & 1023;
    int n   = (b << 10) + hw;
    int k   = idx[n];
    float e = E[d * K + k];
    float x = lat[g];
    out[g]  = e;                                   // straight-through forward value
    float df = e - x;
    float s  = df * df;
    #pragma unroll
    for (int m = 16; m >= 1; m >>= 1) s += __shfl_xor(s, m, 32);
    // loss = q_latent + BETA*e_latent = 1.25 * mean((q-x)^2)
    if ((threadIdx.x & 31) == 0) atomicAdd(loss, s * (1.25f / 8388608.0f));
}

extern "C" void kernel_launch(void* const* d_in, const int* in_sizes, int n_in,
                              void* d_out, int out_size, void* d_ws, size_t ws_size,
                              hipStream_t stream) {
    const float* lat = (const float*)d_in[0];      // [32,256,32,32]
    const float* emb = (const float*)d_in[1];      // [256,8192]

    float* out  = (float*)d_out;                   // [32,256,32,32]
    float* loss = out + TOTAL;                     // scalar
    int*   idxO = (int*)(out + TOTAL + 1);         // [32768] indices (int32 bits)

    unsigned short* A  = (unsigned short*)d_ws;            // [N, D] bf16  (16 MB)
    unsigned short* ET = A + (size_t)NROWS * D;            // [K, D] bf16  ( 4 MB)
    float*          en = (float*)(ET + (size_t)K * D);     // [K] f32      (32 KB)

    hipMemsetAsync(loss, 0, sizeof(float), stream);
    prep_latents<<<TOTAL / 256, 256, 0, stream>>>(lat, A);
    prep_embed  <<<K,           256, 0, stream>>>(emb, ET, en);
    argmin_kernel<<<NROWS / 256, 256, 0, stream>>>(A, ET, en, idxO);
    gather_loss <<<TOTAL / 256, 256, 0, stream>>>(emb, lat, idxO, out, loss);
}
